// MultiHeadAttention_34084860461479
// MI455X (gfx1250) — matrix operand order repaired and checked by reading, not compiled, not run on hardware
//
#include <hip/hip_runtime.h>
#include <hip/hip_bf16.h>

typedef __attribute__((ext_vector_type(16))) __bf16   v16bf;
typedef __attribute__((ext_vector_type(8)))  float    v8f;
typedef __attribute__((ext_vector_type(4)))  unsigned v4u;
typedef __attribute__((ext_vector_type(8)))  int      v8i;
typedef __attribute__((ext_vector_type(4)))  int      v4i;

union Frag16 { v16bf v; unsigned u[8]; };

__device__ __forceinline__ v8f zero8() {
    v8f z = {0.f,0.f,0.f,0.f,0.f,0.f,0.f,0.f};
    return z;
}

// A-matrix fragment (16x32 bf16), base points at row-major [16][pitch] tile.
// Lane L<16: M=L, K in {0..7, 16..23}; lane>=16: M=L-16, K in {8..15, 24..31}.
__device__ __forceinline__ v16bf load_frag_a(const __bf16* base, int pitch, int lane) {
    int row = lane & 15;
    int kb  = (lane >> 4) << 3;           // 0 or 8
    const __bf16* p = base + row * pitch + kb;
    Frag16 f;
#pragma unroll
    for (int i = 0; i < 4; ++i) {
        f.u[i]     = *(const unsigned*)(p + 2 * i);        // K = kb .. kb+7
        f.u[4 + i] = *(const unsigned*)(p + 16 + 2 * i);   // K = kb+16 .. kb+23
    }
    return f.v;
}

// B-matrix fragment (32x16 bf16) loaded from B^T storage: row-major [16(N)][pitch(K)].
// Lane L<16: N=L, K in {0..7, 16..23}; lane>=16: N=L-16, K in {8..15, 24..31}.
__device__ __forceinline__ v16bf load_frag_b(const __bf16* base, int pitch, int lane) {
    int col = lane & 15;
    int kb  = (lane >> 4) << 3;           // 0 or 8
    const __bf16* p = base + col * pitch + kb;
    Frag16 f;
#pragma unroll
    for (int i = 0; i < 4; ++i) {
        f.u[i]     = *(const unsigned*)(p + 2 * i);
        f.u[4 + i] = *(const unsigned*)(p + 16 + 2 * i);
    }
    return f.v;
}

// ---------------------------------------------------------------------------
// Tensor Data Mover: issue a 2-D tile load (global -> LDS) per ISA 8.3/8.4.
// d0flags carries data_size + pad config; groups 2/3 zero (2-D tensor).
// ---------------------------------------------------------------------------
__device__ __forceinline__ void tdm_load_2d(
    unsigned d0flags, unsigned lds_byte, unsigned long long gbytes,
    unsigned tdim0, unsigned tdim1, unsigned tile0, unsigned tile1,
    unsigned stride0)
{
    v4u g0;
    g0[0] = 1u;                                       // count=1 (valid load D#)
    g0[1] = lds_byte;                                 // lds_addr [63:32]
    g0[2] = (unsigned)gbytes;                         // global_addr low
    g0[3] = (unsigned)(gbytes >> 32) | 0x80000000u;   // addr high | type=2 [127:126]
    v8i g1;
    g1[0] = (int)d0flags;                             // mask/data_size/pad cfg
    g1[1] = (int)(tdim0 << 16);                       // tensor_dim0 lo16 @ [63:48]
    g1[2] = (int)((tdim0 >> 16) | (tdim1 << 16));     // dim0 hi16 | dim1 lo16
    g1[3] = (int)((tdim1 >> 16) | (tile0 << 16));     // dim1 hi16 | tile_dim0
    g1[4] = (int)(tile1 & 0xFFFFu);                   // tile_dim1 ; tile_dim2=0
    g1[5] = (int)stride0;                             // tensor_dim0_stride lo32
    g1[6] = 0;                                        // stride hi16 | dim1_stride
    g1[7] = 0;
    v4i z4 = {0, 0, 0, 0};
#if defined(__clang_major__) && (__clang_major__ >= 23)
    v8i z8 = {0, 0, 0, 0, 0, 0, 0, 0};
    __builtin_amdgcn_tensor_load_to_lds(g0, g1, z4, z4, z8, 0);
#else
    __builtin_amdgcn_tensor_load_to_lds(g0, g1, z4, z4, 0);
#endif
}

// d0 flag words: data_size=2B (code 1 @ [17:16]) | pad_enable(bit20) |
// pad_interval @ [24:22] | pad_amount @ [31:25]
// K tile rows are 32 dwords -> interval code 4, pad 2 dwords -> amount code 1
#define TDM_D0_K 0x03110000u
// V tile rows are 16 dwords -> interval code 3, pad 2 dwords -> amount code 1
#define TDM_D0_V 0x02D10000u

// ---------------------------------------------------------------------------
// Kernel 1: projection GEMM  out = X * W^T + b, fp32 in -> bf16 out
// Block tile 128(M) x 64(N), K-chunks of 32, 256 threads = 8 waves (4M x 2N),
// each wave computes a 32x32 tile (2x2 WMMA frags).
// layout 0: out[B,H,S,D]  (token-major heads)    layout 1: out[B,H,D,S] (V^T)
// ---------------------------------------------------------------------------
__global__ __launch_bounds__(256) void proj_gemm_bf16(
    const float* __restrict__ X, const float* __restrict__ W,
    const float* __restrict__ bias, __bf16* __restrict__ out,
    int Ntok, int K, int N, int layout)
{
    const int PITCH = 34;                 // 32 + 2 pad (17-dword row stride)
    __shared__ __bf16 lA[128 * 34];
    __shared__ __bf16 lB[64 * 34];

    const int t    = threadIdx.x;
    const int lane = t & 31;
    const int wave = t >> 5;
    const int wm   = wave >> 1;           // 0..3
    const int wn   = wave & 1;            // 0..1
    const int m0   = blockIdx.x * 128;
    const int n0   = blockIdx.y * 64;

    v8f c[2][2];
#pragma unroll
    for (int i = 0; i < 2; ++i)
#pragma unroll
        for (int j = 0; j < 2; ++j) c[i][j] = zero8();

    for (int k0 = 0; k0 < K; k0 += 32) {
        // stage A: 128x32 fp32 -> bf16 ; 16 floats / thread
        {
            int row = t >> 1, koff = (t & 1) * 16;
            const float4* src = (const float4*)(X + (size_t)(m0 + row) * K + k0 + koff);
#pragma unroll
            for (int i = 0; i < 4; ++i) {
                float4 f = src[i];
                __bf16* d = &lA[row * PITCH + koff + i * 4];
                d[0] = (__bf16)f.x; d[1] = (__bf16)f.y;
                d[2] = (__bf16)f.z; d[3] = (__bf16)f.w;
            }
            if (k0 + 32 < K)
                __builtin_prefetch(X + (size_t)(m0 + row) * K + k0 + 32 + koff, 0, 1);
        }
        // stage B (W rows are already the K-contiguous B^T rows): 64x32
        {
            int row = t >> 2, koff = (t & 3) * 8;
            const float4* src = (const float4*)(W + (size_t)(n0 + row) * K + k0 + koff);
#pragma unroll
            for (int i = 0; i < 2; ++i) {
                float4 f = src[i];
                __bf16* d = &lB[row * PITCH + koff + i * 4];
                d[0] = (__bf16)f.x; d[1] = (__bf16)f.y;
                d[2] = (__bf16)f.z; d[3] = (__bf16)f.w;
            }
        }
        __syncthreads();

        v16bf a0 = load_frag_a(&lA[(wm * 32 +  0) * PITCH], PITCH, lane);
        v16bf a1 = load_frag_a(&lA[(wm * 32 + 16) * PITCH], PITCH, lane);
        v16bf b0 = load_frag_b(&lB[(wn * 32 +  0) * PITCH], PITCH, lane);
        v16bf b1 = load_frag_b(&lB[(wn * 32 + 16) * PITCH], PITCH, lane);

        c[0][0] = __builtin_amdgcn_wmma_f32_16x16x32_bf16(false, a0, false, b0, (short)0, c[0][0], false, false);
        c[0][1] = __builtin_amdgcn_wmma_f32_16x16x32_bf16(false, a0, false, b1, (short)0, c[0][1], false, false);
        c[1][0] = __builtin_amdgcn_wmma_f32_16x16x32_bf16(false, a1, false, b0, (short)0, c[1][0], false, false);
        c[1][1] = __builtin_amdgcn_wmma_f32_16x16x32_bf16(false, a1, false, b1, (short)0, c[1][1], false, false);
        __syncthreads();
    }

    // epilogue: bias + bf16 store with head-major layout mapping
#pragma unroll
    for (int mi = 0; mi < 2; ++mi)
#pragma unroll
        for (int ni = 0; ni < 2; ++ni) {
            int n  = n0 + wn * 32 + ni * 16 + (lane & 15);
            float be = bias[n];
            int h = n >> 6, dd = n & 63;
#pragma unroll
            for (int r = 0; r < 8; ++r) {
                int m  = m0 + wm * 32 + mi * 16 + r + ((lane >> 4) << 3);
                int bb = m >> 11, ss = m & 2047;       // token -> (batch, seq)
                float val = c[mi][ni][r] + be;
                size_t idx;
                if (layout == 0) idx = (((size_t)(bb * 16 + h) * 2048) + ss) * 64 + dd;
                else             idx = (((size_t)(bb * 16 + h) * 64) + dd) * 2048 + ss;
                out[idx] = (__bf16)val;
            }
        }
}

// ---------------------------------------------------------------------------
// Kernel 2: flash attention with TDM double-buffered K/V staging.
// Grid (S/64, H, B), block 128 = 4 waves; each wave owns a 16-query tile.
// Wave 0 issues tensor_load_to_lds for chunk i+1 while all waves compute on
// chunk i; s_wait_tensorcnt + workgroup barrier order the hand-off.
// Q,K layout [B,H,S,D]; V layout [B,H,D,S]; output [B,S,H,D] bf16.
// ---------------------------------------------------------------------------
// LDS plan (element offsets inside one __shared__ block => compile-time
// LDS byte addresses for the TDM descriptors):
//   K bufs : 2 x [32 rows x 68 halves]   (64 data + 4 pad = 32+2 dwords/row)
//   V bufs : 2 x [64 rows x 36 halves]   (32 data + 4 pad = 16+2 dwords/row)
//   P tiles: 4 x [16 rows x 36 halves]
#define LK_ELEMS (32 * 68)
#define LV_ELEMS (64 * 36)
#define LP_ELEMS (16 * 36)
#define LK_OFF(buf) ((buf) * LK_ELEMS)
#define LV_OFF(buf) (2 * LK_ELEMS + (buf) * LV_ELEMS)
#define LP_OFF(w)   (2 * LK_ELEMS + 2 * LV_ELEMS + (w) * LP_ELEMS)
#define SMEM_ELEMS  (2 * LK_ELEMS + 2 * LV_ELEMS + 4 * LP_ELEMS)

__global__ __launch_bounds__(128) void flash_attn_kernel(
    const __bf16* __restrict__ Q, const __bf16* __restrict__ Kh,
    const __bf16* __restrict__ Vt, __bf16* __restrict__ O)
{
    const int S = 2048, D = 64, H = 16;
    const int KP = 68, VP = 36, PP = 36;  // padded LDS pitches (halves)
    __shared__ __bf16 smem[SMEM_ELEMS];

    const int t    = threadIdx.x;
    const int lane = t & 31;
    const int wave = t >> 5;
    const int h    = blockIdx.y;
    const int b    = blockIdx.z;
    const int qbase = (blockIdx.x * 4 + wave) * 16;

    const __bf16* Qbh = Q  + (((size_t)b * H + h) * S) * D;
    const __bf16* Kbh = Kh + (((size_t)b * H + h) * S) * D;
    const __bf16* Vbh = Vt + (((size_t)b * H + h) * D) * S;

    v16bf aq[2];
    aq[0] = load_frag_a(Qbh + (size_t)qbase * D +  0, D, lane);
    aq[1] = load_frag_a(Qbh + (size_t)qbase * D + 32, D, lane);

    v8f o[4];
#pragma unroll
    for (int i = 0; i < 4; ++i) o[i] = zero8();
    float mrun[8], lrun[8];
#pragma unroll
    for (int r = 0; r < 8; ++r) { mrun[r] = -1e30f; lrun[r] = 0.f; }

    const int NCH = S / 32;               // 64 key chunks

    // prologue: DMA chunk 0 into buffer 0
    if (wave == 0) {
        // K tile: tensor [S x 64] halves, tile 64(w) x 32(h)
        tdm_load_2d(TDM_D0_K, LK_OFF(0) * 2,
                    (unsigned long long)(uintptr_t)Kbh,
                    /*tdim0=*/64, /*tdim1=*/S, /*tile0=*/64, /*tile1=*/32,
                    /*stride0=*/64);
        // V^T tile: tensor [64 x S] halves, tile 32(w) x 64(h)
        tdm_load_2d(TDM_D0_V, LV_OFF(0) * 2,
                    (unsigned long long)(uintptr_t)Vbh,
                    /*tdim0=*/S, /*tdim1=*/64, /*tile0=*/32, /*tile1=*/64,
                    /*stride0=*/S);
    }

    for (int ch = 0; ch < NCH; ++ch) {
        const int buf = ch & 1;
        const int s0  = ch * 32;
        if (wave == 0) {
            if (ch + 1 < NCH) {
                const int sn = s0 + 32;
                tdm_load_2d(TDM_D0_K, LK_OFF(buf ^ 1) * 2,
                            (unsigned long long)(uintptr_t)(Kbh + (size_t)sn * D),
                            64, S, 64, 32, 64);
                tdm_load_2d(TDM_D0_V, LV_OFF(buf ^ 1) * 2,
                            (unsigned long long)(uintptr_t)(Vbh + sn),
                            S, 64, 32, 64, S);
                __builtin_amdgcn_s_wait_tensorcnt(2);  // chunk ch landed
            } else {
                __builtin_amdgcn_s_wait_tensorcnt(0);
            }
        }
        __syncthreads();                   // release all waves onto buffer ch

        const __bf16* lK = smem + LK_OFF(buf);
        const __bf16* lV = smem + LV_OFF(buf);

        // S = Q * K^T for 32 keys (two 16-key column tiles), accumulate over d
        v8f sf0 = zero8(), sf1 = zero8();
#pragma unroll
        for (int cchunk = 0; cchunk < 2; ++cchunk) {
            v16bf bk0 = load_frag_b(lK + 0 * 16 * KP + cchunk * 32, KP, lane);
            v16bf bk1 = load_frag_b(lK + 1 * 16 * KP + cchunk * 32, KP, lane);
            sf0 = __builtin_amdgcn_wmma_f32_16x16x32_bf16(false, aq[cchunk], false, bk0, (short)0, sf0, false, false);
            sf1 = __builtin_amdgcn_wmma_f32_16x16x32_bf16(false, aq[cchunk], false, bk1, (short)0, sf1, false, false);
        }

        // online softmax; row q = r + 8*(lane>=16), its 16 scores live across
        // the 16-lane half -> shfl_xor(1,2,4,8) reductions
        float p0[8], p1[8];
#pragma unroll
        for (int r = 0; r < 8; ++r) {
            float v = fmaxf(sf0[r], sf1[r]);
            v = fmaxf(v, __shfl_xor(v, 1, 32));
            v = fmaxf(v, __shfl_xor(v, 2, 32));
            v = fmaxf(v, __shfl_xor(v, 4, 32));
            v = fmaxf(v, __shfl_xor(v, 8, 32));
            float mnew  = fmaxf(mrun[r], v);
            float alpha = __expf(mrun[r] - mnew);
            mrun[r] = mnew;
            p0[r] = __expf(sf0[r] - mnew);
            p1[r] = __expf(sf1[r] - mnew);
            float rs = p0[r] + p1[r];
            rs += __shfl_xor(rs, 1, 32);
            rs += __shfl_xor(rs, 2, 32);
            rs += __shfl_xor(rs, 4, 32);
            rs += __shfl_xor(rs, 8, 32);
            lrun[r] = lrun[r] * alpha + rs;
#pragma unroll
            for (int dt = 0; dt < 4; ++dt) o[dt][r] *= alpha;
        }

        // write P (C layout -> A layout) through this wave's private LDS tile
        __bf16* lP = smem + LP_OFF(wave);
        {
            int qrow = (lane >> 4) << 3;
            int kc   = lane & 15;
#pragma unroll
            for (int r = 0; r < 8; ++r) {
                lP[(qrow + r) * PP + kc]      = (__bf16)p0[r];
                lP[(qrow + r) * PP + kc + 16] = (__bf16)p1[r];
            }
        }
        v16bf ap = load_frag_a(lP, PP, lane);   // same-wave DS RAW, HW-ordered

        // O += P * V  (4 d-tiles of 16)
#pragma unroll
        for (int dt = 0; dt < 4; ++dt) {
            v16bf bv = load_frag_b(lV + dt * 16 * VP, VP, lane);
            o[dt] = __builtin_amdgcn_wmma_f32_16x16x32_bf16(false, ap, false, bv, (short)0, o[dt], false, false);
        }
        __syncthreads();                   // everyone done with buffer ch
    }

    // normalize and store attn output in [B,S,H,D] bf16
    {
        int dlo  = lane & 15;
        int qrow = (lane >> 4) << 3;
#pragma unroll
        for (int dt = 0; dt < 4; ++dt) {
            int d = dt * 16 + dlo;
#pragma unroll
            for (int r = 0; r < 8; ++r) {
                int sq = qbase + qrow + r;
                float val = o[dt][r] / lrun[r];
                O[(((size_t)b * S + sq) * H + h) * D + d] = (__bf16)val;
            }
        }
    }
}

// ---------------------------------------------------------------------------
// Kernel 3: output projection  out_f32 = attn_bf16 * Wo^T + bo
// Same tiling as kernel 1; A already bf16.
// ---------------------------------------------------------------------------
__global__ __launch_bounds__(256) void out_proj_gemm(
    const __bf16* __restrict__ A, const float* __restrict__ W,
    const float* __restrict__ bias, float* __restrict__ out,
    int Ntok, int K, int N)
{
    const int PITCH = 34;
    __shared__ __bf16 lA[128 * 34];
    __shared__ __bf16 lB[64 * 34];

    const int t    = threadIdx.x;
    const int lane = t & 31;
    const int wave = t >> 5;
    const int wm   = wave >> 1;
    const int wn   = wave & 1;
    const int m0   = blockIdx.x * 128;
    const int n0   = blockIdx.y * 64;

    v8f c[2][2];
#pragma unroll
    for (int i = 0; i < 2; ++i)
#pragma unroll
        for (int j = 0; j < 2; ++j) c[i][j] = zero8();

    for (int k0 = 0; k0 < K; k0 += 32) {
        {   // stage A (bf16 copy): 128x32, 16 halves / thread
            int row = t >> 1, koff = (t & 1) * 16;
            const unsigned* src = (const unsigned*)(A + (size_t)(m0 + row) * K + k0 + koff);
            unsigned* dst = (unsigned*)&lA[row * PITCH + koff];
#pragma unroll
            for (int i = 0; i < 8; ++i) dst[i] = src[i];
        }
        {   // stage B: fp32 W -> bf16
            int row = t >> 2, koff = (t & 3) * 8;
            const float4* src = (const float4*)(W + (size_t)(n0 + row) * K + k0 + koff);
#pragma unroll
            for (int i = 0; i < 2; ++i) {
                float4 f = src[i];
                __bf16* d = &lB[row * PITCH + koff + i * 4];
                d[0] = (__bf16)f.x; d[1] = (__bf16)f.y;
                d[2] = (__bf16)f.z; d[3] = (__bf16)f.w;
            }
        }
        __syncthreads();

        v16bf a0 = load_frag_a(&lA[(wm * 32 +  0) * PITCH], PITCH, lane);
        v16bf a1 = load_frag_a(&lA[(wm * 32 + 16) * PITCH], PITCH, lane);
        v16bf b0 = load_frag_b(&lB[(wn * 32 +  0) * PITCH], PITCH, lane);
        v16bf b1 = load_frag_b(&lB[(wn * 32 + 16) * PITCH], PITCH, lane);

        c[0][0] = __builtin_amdgcn_wmma_f32_16x16x32_bf16(false, a0, false, b0, (short)0, c[0][0], false, false);
        c[0][1] = __builtin_amdgcn_wmma_f32_16x16x32_bf16(false, a0, false, b1, (short)0, c[0][1], false, false);
        c[1][0] = __builtin_amdgcn_wmma_f32_16x16x32_bf16(false, a1, false, b0, (short)0, c[1][0], false, false);
        c[1][1] = __builtin_amdgcn_wmma_f32_16x16x32_bf16(false, a1, false, b1, (short)0, c[1][1], false, false);
        __syncthreads();
    }

#pragma unroll
    for (int mi = 0; mi < 2; ++mi)
#pragma unroll
        for (int ni = 0; ni < 2; ++ni) {
            int n = n0 + wn * 32 + ni * 16 + (lane & 15);
            float be = bias[n];
#pragma unroll
            for (int r = 0; r < 8; ++r) {
                int m = m0 + wm * 32 + mi * 16 + r + ((lane >> 4) << 3);
                out[(size_t)m * N + n] = c[mi][ni][r] + be;
            }
        }
}

// ---------------------------------------------------------------------------
extern "C" void kernel_launch(void* const* d_in, const int* in_sizes, int n_in,
                              void* d_out, int out_size, void* d_ws, size_t ws_size,
                              hipStream_t stream) {
    const float* q  = (const float*)d_in[0];
    const float* k  = (const float*)d_in[1];
    const float* v  = (const float*)d_in[2];
    const float* Wq = (const float*)d_in[3];
    const float* bq = (const float*)d_in[4];
    const float* Wk = (const float*)d_in[5];
    const float* bk = (const float*)d_in[6];
    const float* Wv = (const float*)d_in[7];
    const float* bv = (const float*)d_in[8];
    const float* Wo = (const float*)d_in[9];
    const float* bo = (const float*)d_in[10];

    const size_t SZ = (size_t)2 * 16 * 2048 * 64;  // B*H*S*D = 4,194,304 elems
    __bf16* qh   = (__bf16*)d_ws;       // [B,H,S,D]
    __bf16* kh   = qh + SZ;             // [B,H,S,D]
    __bf16* vt   = kh + SZ;             // [B,H,D,S]
    __bf16* attn = vt + SZ;             // [B,S,H,D]

    dim3 gproj(32, 16);                 // 4096/128 x 1024/64
    proj_gemm_bf16<<<gproj, 256, 0, stream>>>(q, Wq, bq, qh, 4096, 1024, 1024, 0);
    proj_gemm_bf16<<<gproj, 256, 0, stream>>>(k, Wk, bk, kh, 4096, 1024, 1024, 0);
    proj_gemm_bf16<<<gproj, 256, 0, stream>>>(v, Wv, bv, vt, 4096, 1024, 1024, 1);

    flash_attn_kernel<<<dim3(32, 16, 2), 128, 0, stream>>>(qh, kh, vt, attn);

    out_proj_gemm<<<gproj, 256, 0, stream>>>(attn, Wo, bo, (float*)d_out, 4096, 1024, 1024);
}
